// ContextAwareDualVQ_24902220382466
// MI455X (gfx1250) — matrix operand order repaired and checked by reading, not compiled, not run on hardware
//
#include <hip/hip_runtime.h>

typedef __attribute__((ext_vector_type(16))) __bf16 v16bf;
typedef __attribute__((ext_vector_type(8)))  float  v8f;

#define LAT  128
#define DIM  256
#define NTOK 262144

// ---------------------------------------------------------------------------
// pack kernel: cb -> bf16 (row major), W[256,K] -> W^T bf16 [K,256]
// run once per launch; tables are L2-resident and reused by every block.
// ---------------------------------------------------------------------------
__global__ void pack_b_kernel(const float* __restrict__ cb, const float* __restrict__ Wm,
                              __bf16* __restrict__ cbbf, __bf16* __restrict__ wtbf, int K) {
  const int i = blockIdx.x * blockDim.x + threadIdx.x;
  if (i < K * DIM) {
    cbbf[i] = (__bf16)cb[i];
    const int n = i >> 8;       // code index
    const int k = i & 255;      // reduce-dim index
    wtbf[i] = (__bf16)Wm[(size_t)k * K + n];   // W^T[n][k]
  }
}

// ---------------------------------------------------------------------------
// main fused VQ kernel: distances+logits via WMMA, softmax/argmin, gather,
// loss SSE, and scatter stats. One 16-token tile per wave.
// ---------------------------------------------------------------------------
template<int K, int WAVES>
__global__ __launch_bounds__(WAVES * 32)
void vq_main(const float* __restrict__ z_re, const float* __restrict__ z_im,
             const float* __restrict__ cb,
             const __bf16* __restrict__ cbbf, const __bf16* __restrict__ wtbf,
             const float* __restrict__ bvec, const float* __restrict__ adj,
             const int*   __restrict__ prev, const float* __restrict__ cbnorm,
             float* __restrict__ zq_out, float* __restrict__ idxf_out,
             int*   __restrict__ idxi_out, float* __restrict__ counts,
             float* __restrict__ embed, float* __restrict__ sse)
{
  __shared__ float dbuf[WAVES][16][K];   // dot(zf, cb[n])
  __shared__ float lbuf[WAVES][16][K];   // logits zf@W
  __shared__ int   idxs[WAVES][16];

  const int wave = threadIdx.x >> 5;
  const int lane = threadIdx.x & 31;
  const int l    = lane & 15;
  const int hf   = lane >> 4;                       // lane half (A/B K-interleave)
  const int m0   = (blockIdx.x * WAVES + wave) * 16;
  const int arow = m0 + l;                          // token row this lane feeds

  // ---- A fragments: zf = [z_re | z_im], 8 chunks of 32 along reduce dim ----
  // 16-bit A 16x32 layout used here: lanes 0-15 hold K k0..k0+15 (contiguous, same as B),
  // lanes 16-31 hold K k0+16..k0+31.
  v16bf afrag[8];
#pragma unroll
  for (int kc = 0; kc < 8; ++kc) {
    const int k0 = kc * 32;
    const float* src = (k0 < LAT) ? (z_re + (size_t)arow * LAT + k0)
                                  : (z_im + (size_t)arow * LAT + (k0 - LAT));
    const float* p = src + hf * 16;
    v16bf a;
#pragma unroll
    for (int e = 0; e < 8; ++e) { a[e] = (__bf16)p[e]; a[8 + e] = (__bf16)p[8 + e]; }
    afrag[kc] = a;
  }

  // ---- dot products vs codebook: per 16-column tile, 8 WMMA k-steps --------
  // B 32x16 bf16: lane l holds column n0+l; lanes 0-15 K=k0..k0+15,
  // lanes 16-31 K=k0+16..k0+31 -> one contiguous 32B run of the packed row.
  for (int nt = 0; nt < K / 16; ++nt) {
    const int n0 = nt * 16;
    v8f acc = {0.f, 0.f, 0.f, 0.f, 0.f, 0.f, 0.f, 0.f};
#pragma unroll
    for (int kc = 0; kc < 8; ++kc) {
      const __bf16* p = cbbf + (size_t)(n0 + l) * DIM + kc * 32 + hf * 16;
      const v16bf b = *(const v16bf*)p;            // 32B aligned, 2x b128
      acc = __builtin_amdgcn_wmma_f32_16x16x32_bf16(false, afrag[kc], false, b,
                                                    (short)0, acc, false, false);
    }
#pragma unroll
    for (int v = 0; v < 8; ++v)
      dbuf[wave][v + hf * 8][n0 + l] = acc[v];
  }

  // ---- context logits: zf @ W via packed W^T ------------------------------
  for (int nt = 0; nt < K / 16; ++nt) {
    const int n0 = nt * 16;
    v8f acc = {0.f, 0.f, 0.f, 0.f, 0.f, 0.f, 0.f, 0.f};
#pragma unroll
    for (int kc = 0; kc < 8; ++kc) {
      const __bf16* p = wtbf + (size_t)(n0 + l) * DIM + kc * 32 + hf * 16;
      const v16bf b = *(const v16bf*)p;
      acc = __builtin_amdgcn_wmma_f32_16x16x32_bf16(false, afrag[kc], false, b,
                                                    (short)0, acc, false, false);
    }
#pragma unroll
    for (int v = 0; v < 8; ++v)
      lbuf[wave][v + hf * 8][n0 + l] = acc[v];
  }

  __syncthreads();

  // ---- per-row epilogue: softmax over logits, history bias, argmin ---------
  if (lane < 16) {
    const int t  = m0 + lane;
    const int pv = prev[t];
    float mx = -3.4e38f;
    for (int n = 0; n < K; ++n) {
      const float lg = lbuf[wave][lane][n] + bvec[n];
      mx = fmaxf(mx, lg);
    }
    float se = 0.f;
    for (int n = 0; n < K; ++n)
      se += expf(lbuf[wave][lane][n] + bvec[n] - mx);
    const float inv = 1.0f / se;
    float best = 3.4e38f; int bi = 0;
    for (int n = 0; n < K; ++n) {
      const float dot  = dbuf[wave][lane][n];
      const float sm   = expf(lbuf[wave][lane][n] + bvec[n] - mx) * inv;
      const float hist = 1.0f / (1.0f + expf(-adj[(size_t)pv * K + n]));
      // ||zf||^2 is constant across n -> dropped from argmin
      const float s = cbnorm[n] - 2.0f * dot - 0.8f * hist - 2.0f * sm;
      if (s < best) { best = s; bi = n; }   // first-min tie-break like argmin
    }
    idxs[wave][lane] = bi;
    idxi_out[t] = bi;
    idxf_out[t] = (float)bi;
    atomicAdd(&counts[bi], 1.0f);
  }
  __syncthreads();

  // ---- gather zq = cb[idx] (f32), SSE for loss, scatter embed_sum ----------
  float lsum = 0.f;
  for (int r = 0; r < 16; ++r) {
    const int t  = m0 + r;
    const int bi = idxs[wave][r];
    const float* crow = cb + (size_t)bi * DIM;
#pragma unroll
    for (int e = 0; e < 8; ++e) {
      const int dI = lane * 8 + e;                  // 0..255 across the wave
      const float zq  = crow[dI];
      const float zfv = (dI < LAT) ? z_re[(size_t)t * LAT + dI]
                                   : z_im[(size_t)t * LAT + dI - LAT];
      // streaming store: keep the 192MB L2 for cb/W/adj, not write-once zq
      __builtin_nontemporal_store(zq, &zq_out[(size_t)t * DIM + dI]);
      const float df = zq - zfv;
      lsum += df * df;
      atomicAdd(&embed[(size_t)bi * DIM + dI], zfv);
    }
  }
#pragma unroll
  for (int off = 16; off > 0; off >>= 1) lsum += __shfl_xor(lsum, off, 32);
  if (lane == 0) atomicAdd(sse, lsum);
}

// ---------------------------------------------------------------------------
// small kernels
// ---------------------------------------------------------------------------
__global__ void cbnorm_kernel(const float* __restrict__ cb, float* __restrict__ out, int K) {
  const int k = blockIdx.x * blockDim.x + threadIdx.x;
  if (k < K) {
    float s = 0.f;
    for (int d = 0; d < DIM; ++d) { const float v = cb[(size_t)k * DIM + d]; s += v * v; }
    out[k] = s;
  }
}

__global__ void copy_kernel(const float* __restrict__ src, float* __restrict__ dst, int n) {
  const int i = blockIdx.x * blockDim.x + threadIdx.x;
  if (i < n) dst[i] = src[i];
}

__global__ void adj_scatter(const float* __restrict__ adj_in, float* __restrict__ adj_out,
                            const int* __restrict__ prev, const int* __restrict__ idx,
                            int K, int n) {
  const int t = blockIdx.x * blockDim.x + threadIdx.x;
  if (t < n) {
    const int p = prev[t], i = idx[t];
    // last-write-wins, matching the reference's indexed-assign semantics
    adj_out[(size_t)p * K + i] = adj_in[(size_t)p * K + i] * 0.995f + 1.0f;
  }
}

template<int K>
__global__ void finalize_stats(const float* __restrict__ cl,     const float* __restrict__ avg,
                               const float* __restrict__ counts, const float* __restrict__ embed,
                               float* __restrict__ cl_out, float* __restrict__ avg_out,
                               float* __restrict__ cb_out)
{
  __shared__ float red[256];
  float local = 0.f;
  for (int k = threadIdx.x; k < K; k += 256) {
    const float c = cl[k] * 0.99f + 0.01f * counts[k];
    cl_out[k] = c;
    local += c;
  }
  red[threadIdx.x] = local;
  __syncthreads();
  for (int s = 128; s > 0; s >>= 1) {
    if ((int)threadIdx.x < s) red[threadIdx.x] += red[threadIdx.x + s];
    __syncthreads();
  }
  const float n = red[0];
  for (int i = threadIdx.x; i < K * DIM; i += 256) {
    const int k = i >> 8;
    const float c = cl[k] * 0.99f + 0.01f * counts[k];      // recompute: no cross-thread dep
    const float a = avg[i] * 0.99f + 0.01f * embed[i];
    avg_out[i] = a;
    const float cs = (c + 1e-6f) / (n + (float)K * 1e-6f) * n;
    cb_out[i] = a / cs;
  }
}

__global__ void finalize_loss(const float* __restrict__ sse, float* __restrict__ loss) {
  // loss = 1.25*mean_syn + 1.25*mean_sem  (sg terms collapse numerically)
  *loss = 1.25f * (sse[0] + sse[1]) / ((float)NTOK * (float)DIM);
}

// ---------------------------------------------------------------------------
// launch
// ---------------------------------------------------------------------------
extern "C" void kernel_launch(void* const* d_in, const int* in_sizes, int n_in,
                              void* d_out, int out_size, void* d_ws, size_t ws_size,
                              hipStream_t stream) {
  const float* z_fast_re = (const float*)d_in[0];
  const float* z_fast_im = (const float*)d_in[1];
  const float* z_slow_re = (const float*)d_in[2];
  const float* z_slow_im = (const float*)d_in[3];
  const float* cb_syn    = (const float*)d_in[4];
  const float* cb_sem    = (const float*)d_in[5];
  const float* W_syn     = (const float*)d_in[6];
  const float* b_syn     = (const float*)d_in[7];
  const float* W_sem     = (const float*)d_in[8];
  const float* b_sem     = (const float*)d_in[9];
  const float* cl_syn    = (const float*)d_in[10];
  const float* avg_syn   = (const float*)d_in[11];
  const float* cl_sem    = (const float*)d_in[12];
  const float* avg_sem   = (const float*)d_in[13];
  const float* adj_syn   = (const float*)d_in[14];
  const float* adj_sem   = (const float*)d_in[15];
  const int*   prev_syn  = (const int*)d_in[16];
  const int*   prev_sem  = (const int*)d_in[17];

  float* O = (float*)d_out;
  float* zq_syn_o  = O;
  float* zq_sem_o  = zq_syn_o + (size_t)NTOK * DIM;
  float* loss_o    = zq_sem_o + (size_t)NTOK * DIM;
  float* idxf_syn  = loss_o + 1;
  float* idxf_sem  = idxf_syn + NTOK;
  float* cb_syn_o  = idxf_sem + NTOK;
  float* cl_syn_o  = cb_syn_o + 64 * DIM;
  float* avg_syn_o = cl_syn_o + 64;
  float* adj_syn_o = avg_syn_o + 64 * DIM;
  float* cb_sem_o  = adj_syn_o + 64 * 64;
  float* cl_sem_o  = cb_sem_o + 128 * DIM;
  float* avg_sem_o = cl_sem_o + 128;
  float* adj_sem_o = avg_sem_o + 128 * DIM;

  float* ws         = (float*)d_ws;
  float* counts_syn = ws;                 //    64
  float* counts_sem = ws + 64;            //   128
  float* embed_syn  = ws + 192;           // 16384
  float* embed_sem  = ws + 16576;         // 32768
  float* sse        = ws + 49344;         //     2  (syn, sem)
  float* cbn_syn    = ws + 49346;         //    64
  float* cbn_sem    = ws + 49410;         //   128
  // bf16 pack region, 32B aligned (float offset 49664 -> byte 198656 = 32*6208)
  __bf16* cbbf_syn  = (__bf16*)(ws + 49664);        // 16384 bf16
  __bf16* wtbf_syn  = cbbf_syn + 64 * DIM;          // 16384 bf16
  __bf16* cbbf_sem  = wtbf_syn + 64 * DIM;          // 32768 bf16
  __bf16* wtbf_sem  = cbbf_sem + 128 * DIM;         // 32768 bf16
  int*    idxi_syn  = (int*)(ws + 49664 + 49152);   // NTOK ints
  int*    idxi_sem  = idxi_syn + NTOK;

  // zero the accumulated stats region (counts/embed/sse) every call
  hipMemsetAsync(d_ws, 0, (size_t)49346 * sizeof(float), stream);

  cbnorm_kernel<<<1, 256, 0, stream>>>(cb_syn, cbn_syn, 64);
  cbnorm_kernel<<<1, 256, 0, stream>>>(cb_sem, cbn_sem, 128);
  pack_b_kernel<<<(64 * DIM + 255) / 256, 256, 0, stream>>>(cb_syn, W_syn, cbbf_syn, wtbf_syn, 64);
  pack_b_kernel<<<(128 * DIM + 255) / 256, 256, 0, stream>>>(cb_sem, W_sem, cbbf_sem, wtbf_sem, 128);

  // syn: K=64, 4 waves/block, 64 tokens/block -> 4096 blocks
  vq_main<64, 4><<<NTOK / 64, 128, 0, stream>>>(
      z_fast_re, z_fast_im, cb_syn, cbbf_syn, wtbf_syn, b_syn, adj_syn, prev_syn, cbn_syn,
      zq_syn_o, idxf_syn, idxi_syn, counts_syn, embed_syn, sse + 0);

  // sem: K=128, 2 waves/block, 32 tokens/block -> 8192 blocks
  vq_main<128, 2><<<NTOK / 32, 64, 0, stream>>>(
      z_slow_re, z_slow_im, cb_sem, cbbf_sem, wtbf_sem, b_sem, adj_sem, prev_sem, cbn_sem,
      zq_sem_o, idxf_sem, idxi_sem, counts_sem, embed_sem, sse + 1);

  // adjacency: copy then scatter (stream order gives copy-before-scatter)
  copy_kernel<<<(64 * 64 + 255) / 256, 256, 0, stream>>>(adj_syn, adj_syn_o, 64 * 64);
  adj_scatter<<<(NTOK + 255) / 256, 256, 0, stream>>>(adj_syn, adj_syn_o, prev_syn, idxi_syn, 64, NTOK);
  copy_kernel<<<(128 * 128 + 255) / 256, 256, 0, stream>>>(adj_sem, adj_sem_o, 128 * 128);
  adj_scatter<<<(NTOK + 255) / 256, 256, 0, stream>>>(adj_sem, adj_sem_o, prev_sem, idxi_sem, 128, NTOK);

  finalize_stats<64><<<1, 256, 0, stream>>>(cl_syn, avg_syn, counts_syn, embed_syn,
                                            cl_syn_o, avg_syn_o, cb_syn_o);
  finalize_stats<128><<<1, 256, 0, stream>>>(cl_sem, avg_sem, counts_sem, embed_sem,
                                             cl_sem_o, avg_sem_o, cb_sem_o);
  finalize_loss<<<1, 1, 0, stream>>>(sse, loss_o);
}
